// MMMO_13400297963981
// MI455X (gfx1250) — matrix operand order repaired and checked by reading, not compiled, not run on hardware
//
#include <hip/hip_runtime.h>
#include <hip/hip_bf16.h>
#include <math.h>

typedef __attribute__((ext_vector_type(16))) __bf16 v16bf;
typedef __attribute__((ext_vector_type(8)))  __bf16 v8bf;
typedef __attribute__((ext_vector_type(8)))  float  v8f;

namespace cfg {
constexpr int MM = 4, N_ITEM = 40000, N_USER = 10000, D = 128, DX = 64;
constexpr int E = 400000, B = 4096;
constexpr int N = N_ITEM + N_USER;                 // 50000
}

__device__ __forceinline__ float lrelu(float x) { return x > 0.f ? x : 0.01f * x; }

__device__ __forceinline__ void atomicMaxF(float* addr, float val) {
  unsigned int* ua = (unsigned int*)addr;
  unsigned int old = *ua;
  while (true) {
    float f = __uint_as_float(old);
    if (f >= val) break;
    unsigned int assumed = old;
    old = atomicCAS(ua, assumed, __float_as_uint(val));
    if (old == assumed) break;
  }
}

// ---------------- utility kernels ----------------

__global__ void k_fill(float* p, int n, float v) {
  int i = blockIdx.x * blockDim.x + threadIdx.x;
  if (i < n) p[i] = v;
}

__global__ void k_f2bf(const float* __restrict__ in, __bf16* __restrict__ out, int n) {
  int i = blockIdx.x * blockDim.x + threadIdx.x;
  if (i < n) out[i] = (__bf16)in[i];
}

// convW is [M][K=128][N=128]; produce bf16 [M][N][K] (so GEMM B is row-per-output-col)
__global__ void k_convW_T(const float* __restrict__ in, __bf16* __restrict__ out, int n_total) {
  int i = blockIdx.x * blockDim.x + threadIdx.x;
  if (i >= n_total) return;
  int m = i >> 14;
  int r = i & 16383;
  int n = r >> 7;
  int k = r & 127;
  out[i] = (__bf16)in[(m << 14) + (k << 7) + n];
}

__global__ void k_scale(const float* __restrict__ in, float* __restrict__ out, int n, float s) {
  int i = blockIdx.x * blockDim.x + threadIdx.x;
  if (i < n) out[i] = in[i] * s;
}

// ---------------- WMMA GEMM with LDS-staged weights ----------------
// C[M, NT*16] = epilogue(A[M,128] @ Bt[NT*16,128]^T)
// Block = 256 threads (8 waves). The whole block async-copies Bt (NT*16 x 128 bf16,
// 16/32 KB) into LDS once (GLOBAL_LOAD_ASYNC_TO_LDS_B128, ASYNCcnt), then each wave
// owns one 16-row M-tile and keeps NT accumulators, reusing its A fragment across
// all NT column tiles (B fragments come from LDS via ds_load_b128).
// MODE 0: tanh(acc + bias[col])
// MODE 1: acc (no bias)
// MODE 2: lrelu(acc + bias[col]) + extra[row,col]           (extra = id_emb)
// MODE 3: lrelu(acc + bias[col] + extra[row,col]); C2 += v  (extra = x_hat, C2 = rep accum)
template <int NT, int MODE>
__global__ void k_wmma_gemm(const __bf16* __restrict__ A, int lda,
                            const __bf16* __restrict__ Bt,
                            const float* __restrict__ bias,
                            const float* __restrict__ extra,
                            float* __restrict__ C,
                            float* __restrict__ C2,
                            int ldc, int Mrows) {
  constexpr int K = 128;
  __shared__ __bf16 sB[NT * 16 * K];               // only LDS object -> offset 0

  // ---- stage Bt into LDS with async global->LDS copies (all 256 threads) ----
  {
    const int tid = threadIdx.x;
#pragma unroll
    for (int i = 0; i < NT; ++i) {
      int chunk = tid + i * 256;                   // 16-byte chunk index
      unsigned lds_off = (unsigned)chunk * 16u;    // LDS byte address (sB at offset 0)
      const __bf16* gp = Bt + (size_t)chunk * 8;   // 8 bf16 = 16 bytes
      asm volatile("global_load_async_to_lds_b128 %0, %1, off"
                   :: "v"(lds_off), "v"(gp)
                   : "memory");
    }
    asm volatile("s_wait_asynccnt 0" ::: "memory");
    __syncthreads();                               // LDS visible to all waves
  }

  // ---- per-wave M-tile (wave-uniform bounds check AFTER the barrier) ----
  const int mtile = blockIdx.x * (blockDim.x >> 5) + (threadIdx.x >> 5);
  if (mtile >= (Mrows >> 4)) return;

  const int lane = threadIdx.x & 31;
  const int hv   = lane >> 4;                      // 16-lane half
  const int r16  = lane & 15;

  const __bf16* Arow = A + (size_t)(mtile * 16 + r16) * lda;

  v8f zero = {0.f, 0.f, 0.f, 0.f, 0.f, 0.f, 0.f, 0.f};
  v8f acc[NT];
#pragma unroll
  for (int nt = 0; nt < NT; ++nt) acc[nt] = zero;

#pragma unroll
  for (int k0 = 0; k0 < K; k0 += 32) {
    __builtin_prefetch(Arow + k0 + 64, 0, 1);      // global_prefetch_b8 on streamed A
    // ISA 16-bit A layout: lanes0-15 -> K[k0..k0+7] & K[k0+16..k0+23]; lanes16-31 +8
    v8bf alo = *(const v8bf*)(Arow + k0 + hv * 8);
    v8bf ahi = *(const v8bf*)(Arow + k0 + 16 + hv * 8);
    v16bf a;
#pragma unroll
    for (int i = 0; i < 8; ++i) { a[i] = alo[i]; a[i + 8] = ahi[i]; }

#pragma unroll
    for (int nt = 0; nt < NT; ++nt) {
      // B layout matches A: lanes0-15 -> K[k0..k0+7] & K[k0+16..k0+23]; lanes16-31 +8
      const __bf16* Bcol = &sB[(nt * 16 + r16) * K + k0 + hv * 8];
      v8bf blo = *(const v8bf*)(Bcol);
      v8bf bhi = *(const v8bf*)(Bcol + 16);
      v16bf b;
#pragma unroll
      for (int i = 0; i < 8; ++i) { b[i] = blo[i]; b[i + 8] = bhi[i]; }
      acc[nt] = __builtin_amdgcn_wmma_f32_16x16x32_bf16(false, a, false, b, (short)0,
                                                        acc[nt], false, false);
    }
  }

  // ---- epilogue ----
  const int row0 = mtile * 16 + hv * 8;            // C/D layout: VGPR r -> M = hv*8 + r
#pragma unroll
  for (int nt = 0; nt < NT; ++nt) {
    int col = nt * 16 + r16;
    float bv = (MODE == 1) ? 0.f : bias[col];
#pragma unroll
    for (int r = 0; r < 8; ++r) {
      int row = row0 + r;
      float v = acc[nt][r];
      if (MODE == 0)      v = tanhf(v + bv);
      else if (MODE == 2) v = lrelu(v + bv) + extra[(size_t)row * ldc + col];
      else if (MODE == 3) v = lrelu(v + bv + extra[(size_t)row * ldc + col]);
      C[(size_t)row * ldc + col] = v;
      if (MODE == 3) C2[(size_t)row * ldc + col] += v;
    }
  }
}

// ---------------- row-wise L2 normalize -> bf16 (one wave32 per 128-dim row) ------

__global__ void k_l2norm_bf(const float* __restrict__ x, __bf16* __restrict__ xb, int rows) {
  int row = blockIdx.x * (blockDim.x >> 5) + (threadIdx.x >> 5);
  if (row >= rows) return;
  int lane = threadIdx.x & 31;
  float4 v = ((const float4*)(x + (size_t)row * 128))[lane];
  float ss = v.x * v.x + v.y * v.y + v.z * v.z + v.w * v.w;
  for (int o = 16; o; o >>= 1) ss += __shfl_xor(ss, o, 32);
  float s = 1.f / fmaxf(sqrtf(ss), 1e-12f);
  __bf16* o = xb + (size_t)row * 128 + lane * 4;
  o[0] = (__bf16)(v.x * s); o[1] = (__bf16)(v.y * s);
  o[2] = (__bf16)(v.z * s); o[3] = (__bf16)(v.w * s);
}

// h = lrelu(l2norm(agg + convb)) -> bf16
__global__ void k_hnorm(const float* __restrict__ agg, const float* __restrict__ cb,
                        __bf16* __restrict__ hb, int rows) {
  int row = blockIdx.x * (blockDim.x >> 5) + (threadIdx.x >> 5);
  if (row >= rows) return;
  int lane = threadIdx.x & 31;
  float4 v = ((const float4*)(agg + (size_t)row * 128))[lane];
  float4 b = ((const float4*)cb)[lane];
  v.x += b.x; v.y += b.y; v.z += b.z; v.w += b.w;
  float ss = v.x * v.x + v.y * v.y + v.z * v.z + v.w * v.w;
  for (int o = 16; o; o >>= 1) ss += __shfl_xor(ss, o, 32);
  float s = 1.f / fmaxf(sqrtf(ss), 1e-12f);
  __bf16* o = hb + (size_t)row * 128 + lane * 4;
  o[0] = (__bf16)lrelu(v.x * s); o[1] = (__bf16)lrelu(v.y * s);
  o[2] = (__bf16)lrelu(v.z * s); o[3] = (__bf16)lrelu(v.w * s);
}

// ---------------- edge phase ----------------

__global__ void k_deg(const int* __restrict__ src, float* __restrict__ deg, int E) {
  int e = blockIdx.x * blockDim.x + threadIdx.x;
  if (e < E) atomicAdd(&deg[src[e]], 1.f);
}

// one wave per edge: ip = dot(xt[dst], lrelu(xt[src])); w = ip*sigmoid(dinv*ip); max into wmax
__global__ void k_edge_logits(const float* __restrict__ xt, const int* __restrict__ src,
                              const int* __restrict__ dst, const float* __restrict__ deg,
                              float* __restrict__ wbuf, float* __restrict__ wmax, int E) {
  int e = blockIdx.x * (blockDim.x >> 5) + (threadIdx.x >> 5);
  if (e >= E) return;
  int lane = threadIdx.x & 31;
  int s = src[e], d = dst[e];
  float4 a = ((const float4*)(xt + (size_t)s * 128))[lane];
  float4 b = ((const float4*)(xt + (size_t)d * 128))[lane];
  float ip = b.x * lrelu(a.x) + b.y * lrelu(a.y) + b.z * lrelu(a.z) + b.w * lrelu(a.w);
  for (int o = 16; o; o >>= 1) ip += __shfl_xor(ip, o, 32);
  if (lane == 0) {
    float dg   = deg[s];
    float dinv = dg > 0.f ? rsqrtf(dg) : 0.f;
    float w    = ip * (1.f / (1.f + expf(-dinv * ip)));
    wbuf[e] = w;
    atomicMaxF(&wmax[d], w);
  }
}

__global__ void k_edge_exp(const int* __restrict__ dst, float* __restrict__ wbuf,
                           const float* __restrict__ wmax, float* __restrict__ esum, int E) {
  int e = blockIdx.x * blockDim.x + threadIdx.x;
  if (e >= E) return;
  int d = dst[e];
  float ev = expf(wbuf[e] - wmax[d]);
  wbuf[e] = ev;
  atomicAdd(&esum[d], ev);
}

// one wave per edge: agg[dst] += xt[src] * att
__global__ void k_edge_scatter(const float* __restrict__ xt, const int* __restrict__ src,
                               const int* __restrict__ dst, const float* __restrict__ wbuf,
                               const float* __restrict__ esum, float* __restrict__ agg, int E) {
  int e = blockIdx.x * (blockDim.x >> 5) + (threadIdx.x >> 5);
  if (e >= E) return;
  int lane = threadIdx.x & 31;
  int s = src[e], d = dst[e];
  float att = wbuf[e] / (esum[d] + 1e-16f);
  float4 a = ((const float4*)(xt + (size_t)s * 128))[lane];
  float* base = agg + (size_t)d * 128 + lane * 4;
  atomicAdd(base + 0, a.x * att);
  atomicAdd(base + 1, a.y * att);
  atomicAdd(base + 2, a.z * att);
  atomicAdd(base + 3, a.w * att);
}

// ---------------- cross-modal attention tail ----------------

// Kmod[m][n] = mean_b out_m[pos_items[b]][n]   (grid = 64 cols)
__global__ void k_kmod(const float* __restrict__ out_m, const int* __restrict__ pos,
                       float* __restrict__ kmod) {
  __shared__ float sh[256];
  int n = blockIdx.x;
  float s = 0.f;
  for (int b = threadIdx.x; b < cfg::B; b += 256) s += out_m[(size_t)pos[b] * 64 + n];
  sh[threadIdx.x] = s;
  __syncthreads();
  for (int o = 128; o; o >>= 1) {
    if ((int)threadIdx.x < o) sh[threadIdx.x] += sh[threadIdx.x + o];
    __syncthreads();
  }
  if (threadIdx.x == 0) kmod[n] = sh[0] * (1.f / cfg::B);
}

__global__ void k_kpvp(const float* __restrict__ kmod, const float* __restrict__ kW,
                       const float* __restrict__ vW, float* __restrict__ kp,
                       float* __restrict__ vp) {
  int n = threadIdx.x;  // 64 threads
  for (int j = 0; j < 4; ++j) {
    float a = 0.f, b = 0.f;
    for (int k = 0; k < 64; ++k) {
      float km = kmod[j * 64 + k];
      a += km * kW[n * 64 + k];
      b += km * vW[n * 64 + k];
    }
    kp[j * 64 + n] = a;
    vp[j * 64 + n] = b;
  }
}

// per-query fused: Q, 4-way softmax attention, scores, price MLP (grid = B, block = 64)
__global__ void k_final(const float* __restrict__ rep, const int* __restrict__ user_nodes,
                        const int* __restrict__ pos_items, const int* __restrict__ neg_items,
                        const float* __restrict__ qW, const float* __restrict__ kp,
                        const float* __restrict__ vp, const float* __restrict__ mpW1,
                        const float* __restrict__ mpb1, const float* __restrict__ mpW2,
                        const float* __restrict__ mpb2, float* __restrict__ pos_out,
                        float* __restrict__ neg_out, float* __restrict__ price_out) {
  __shared__ float su[64], sp[64], sn[64], sq[64], satt[64], shid[64], sl[4];
  int b = blockIdx.x;
  int t = threadIdx.x;
  su[t] = rep[(size_t)user_nodes[b] * 64 + t];
  sp[t] = rep[(size_t)pos_items[b] * 64 + t];
  sn[t] = rep[(size_t)neg_items[b] * 64 + t];
  __syncthreads();
  float q = 0.f;
  for (int k = 0; k < 64; ++k) q += su[k] * qW[t * 64 + k];
  sq[t] = q;
  __syncthreads();
  if (t < 4) {
    float l = 0.f;
    for (int k = 0; k < 64; ++k) l += sq[k] * kp[t * 64 + k];
    sl[t] = l * 0.125f;  // / sqrt(64)
  }
  __syncthreads();
  if (t == 0) {
    float mx = fmaxf(fmaxf(sl[0], sl[1]), fmaxf(sl[2], sl[3]));
    float s = 0.f;
    for (int j = 0; j < 4; ++j) { sl[j] = expf(sl[j] - mx); s += sl[j]; }
    float inv = 1.f / s;
    for (int j = 0; j < 4; ++j) sl[j] *= inv;
  }
  __syncthreads();
  float a = 0.f;
  for (int j = 0; j < 4; ++j) a += sl[j] * vp[j * 64 + t];
  satt[t] = a;
  __syncthreads();
  float h = 0.f;
  for (int k = 0; k < 64; ++k) h += satt[k] * mpW1[t * 128 + k];
  for (int k = 0; k < 64; ++k) h += sp[k] * mpW1[t * 128 + 64 + k];
  shid[t] = lrelu(h + mpb1[t]);
  __syncthreads();
  if (t == 0) {
    float ps = 0.f, ns = 0.f, pr = 0.f;
    for (int k = 0; k < 64; ++k) {
      ps += satt[k] * sp[k];
      ns += satt[k] * sn[k];
      pr += shid[k] * mpW2[k];
    }
    pos_out[b] = ps;
    neg_out[b] = ns;
    price_out[b] = pr + mpb2[0];
  }
}

// ---------------- host launch ----------------

static inline int cdiv(int a, int b) { return (a + b - 1) / b; }

extern "C" void kernel_launch(void* const* d_in, const int* in_sizes, int n_in,
                              void* d_out, int out_size, void* d_ws, size_t ws_size,
                              hipStream_t stream) {
  using namespace cfg;
  (void)in_sizes; (void)n_in; (void)out_size; (void)ws_size;

  const float* feats         = (const float*)d_in[0];
  const float* user_features = (const float*)d_in[1];
  const float* userW         = (const float*)d_in[2];
  const float* userb         = (const float*)d_in[3];
  const float* convW         = (const float*)d_in[4];
  const float* convb         = (const float*)d_in[5];
  const float* lin1W         = (const float*)d_in[6];
  const float* lin1b         = (const float*)d_in[7];
  const float* g1W           = (const float*)d_in[8];
  const float* g1b           = (const float*)d_in[9];
  const float* id_emb        = (const float*)d_in[10];
  const float* qW            = (const float*)d_in[11];
  const float* kW            = (const float*)d_in[12];
  const float* vW            = (const float*)d_in[13];
  const float* mpW1          = (const float*)d_in[14];
  const float* mpb1          = (const float*)d_in[15];
  const float* mpW2          = (const float*)d_in[16];
  const float* mpb2          = (const float*)d_in[17];
  const int*   edge_index    = (const int*)d_in[18];
  const int*   user_nodes    = (const int*)d_in[19];
  const int*   pos_items     = (const int*)d_in[20];
  const int*   neg_items     = (const int*)d_in[21];
  const int*   src = edge_index;
  const int*   dst = edge_index + E;

  // workspace carve-out (256B aligned regions)
  char* w = (char*)d_ws;
  auto alloc = [&](size_t bytes) -> char* {
    char* p = w;
    w += (bytes + 255) & ~(size_t)255;
    return p;
  };
  __bf16* wt_user = (__bf16*)alloc((size_t)MM * 128 * 128 * 2);
  __bf16* wt_conv = (__bf16*)alloc((size_t)MM * 128 * 128 * 2);
  __bf16* wt_lin1 = (__bf16*)alloc((size_t)MM * 64 * 128 * 2);
  __bf16* wt_g1   = (__bf16*)alloc((size_t)MM * 64 * 128 * 2);
  __bf16* uf_bf   = (__bf16*)alloc((size_t)N_USER * 128 * 2);
  float*  x_f     = (float*)alloc((size_t)N * 128 * 4);
  __bf16* x_bf    = (__bf16*)alloc((size_t)N * 128 * 2);
  float*  xt_f    = (float*)alloc((size_t)N * 128 * 4);
  float*  agg     = (float*)alloc((size_t)N * 128 * 4);   // reused as out_m after h
  __bf16* h_bf    = (__bf16*)alloc((size_t)N * 128 * 2);
  float*  xhat    = (float*)alloc((size_t)N * 64 * 4);
  float*  repsum  = (float*)alloc((size_t)N * 64 * 4);
  float*  wbuf    = (float*)alloc((size_t)E * 4);
  float*  deg     = (float*)alloc((size_t)N * 4);
  float*  wmax    = (float*)alloc((size_t)N * 4);
  float*  esum    = (float*)alloc((size_t)N * 4);
  float*  kmod    = (float*)alloc((size_t)MM * 64 * 4);
  float*  kp      = (float*)alloc((size_t)MM * 64 * 4);
  float*  vp      = (float*)alloc((size_t)MM * 64 * 4);
  float*  out_m   = agg;

  float* pos_out   = (float*)d_out;
  float* neg_out   = pos_out + B;
  float* rep_out   = neg_out + B;
  float* price_out = rep_out + (size_t)N * DX;

  // ---- one-time prep ----
  k_f2bf<<<cdiv(N_USER * 128, 256), 256, 0, stream>>>(user_features, uf_bf, N_USER * 128);
  k_f2bf<<<cdiv(MM * 128 * 128, 256), 256, 0, stream>>>(userW, wt_user, MM * 128 * 128);
  k_f2bf<<<cdiv(MM * 64 * 128, 256), 256, 0, stream>>>(lin1W, wt_lin1, MM * 64 * 128);
  k_f2bf<<<cdiv(MM * 64 * 128, 256), 256, 0, stream>>>(g1W, wt_g1, MM * 64 * 128);
  k_convW_T<<<cdiv(MM * 128 * 128, 256), 256, 0, stream>>>(convW, wt_conv, MM * 128 * 128);
  k_fill<<<cdiv(N * 64, 256), 256, 0, stream>>>(repsum, N * 64, 0.f);
  k_fill<<<cdiv(N, 256), 256, 0, stream>>>(deg, N, 0.f);
  k_deg<<<cdiv(E, 256), 256, 0, stream>>>(src, deg, E);

  for (int m = 0; m < MM; ++m) {
    // x rows [0, N_ITEM): feats[m]; rows [N_ITEM, N): tanh(uf @ userW[m]^T + userb[m])
    hipMemcpyAsync(x_f, feats + (size_t)m * N_ITEM * 128, (size_t)N_ITEM * 128 * 4,
                   hipMemcpyDeviceToDevice, stream);
    k_wmma_gemm<8, 0><<<cdiv(N_USER / 16, 8), 256, 0, stream>>>(
        uf_bf, 128, wt_user + (size_t)m * 128 * 128, userb + m * 128, nullptr,
        x_f + (size_t)N_ITEM * 128, nullptr, 128, N_USER);
    k_l2norm_bf<<<cdiv(N, 8), 256, 0, stream>>>(x_f, x_bf, N);

    // xt = x @ convW[m]
    k_wmma_gemm<8, 1><<<cdiv(N / 16, 8), 256, 0, stream>>>(
        x_bf, 128, wt_conv + (size_t)m * 128 * 128, nullptr, nullptr, xt_f, nullptr,
        128, N);

    // edge attention + aggregation
    k_fill<<<cdiv(N * 128, 256), 256, 0, stream>>>(agg, N * 128, 0.f);
    k_fill<<<cdiv(N, 256), 256, 0, stream>>>(wmax, N, -3.402823466e38f);
    k_fill<<<cdiv(N, 256), 256, 0, stream>>>(esum, N, 0.f);
    k_edge_logits<<<cdiv(E, 8), 256, 0, stream>>>(xt_f, src, dst, deg, wbuf, wmax, E);
    k_edge_exp<<<cdiv(E, 256), 256, 0, stream>>>(dst, wbuf, wmax, esum, E);
    k_edge_scatter<<<cdiv(E, 8), 256, 0, stream>>>(xt_f, src, dst, wbuf, esum, agg, E);
    k_hnorm<<<cdiv(N, 8), 256, 0, stream>>>(agg, convb + m * 128, h_bf, N);

    // x_hat = lrelu(x @ lin1W[m]^T + lin1b) + id_emb
    k_wmma_gemm<4, 2><<<cdiv(N / 16, 8), 256, 0, stream>>>(
        x_bf, 128, wt_lin1 + (size_t)m * 64 * 128, lin1b + m * 64, id_emb, xhat, nullptr,
        64, N);
    // out_m = lrelu(h @ g1W[m]^T + g1b + x_hat); repsum += out_m
    k_wmma_gemm<4, 3><<<cdiv(N / 16, 8), 256, 0, stream>>>(
        h_bf, 128, wt_g1 + (size_t)m * 64 * 128, g1b + m * 64, xhat, out_m, repsum,
        64, N);

    k_kmod<<<64, 256, 0, stream>>>(out_m, pos_items, kmod + m * 64);
  }

  // rep = repsum / M
  k_scale<<<cdiv(N * 64, 256), 256, 0, stream>>>(repsum, rep_out, N * 64, 1.f / MM);
  // Kp / Vp from modality means
  k_kpvp<<<1, 64, 0, stream>>>(kmod, kW, vW, kp, vp);
  // fused attention + scores + price MLP
  k_final<<<B, 64, 0, stream>>>(rep_out, user_nodes, pos_items, neg_items, qW, kp, vp,
                                mpW1, mpb1, mpW2, mpb2, pos_out, neg_out, price_out);
}
